// SparseMoE_22351009809010
// MI455X (gfx1250) — matrix operand order repaired and checked by reading, not compiled, not run on hardware
//
#include <hip/hip_runtime.h>
#include <math.h>

#define TOK   8192
#define DIM   1024
#define HID   4096
#define NE    8
#define TOPK  2
#define CAP   2150
#define CPAD  2176   // 17 * 128

typedef __attribute__((ext_vector_type(16))) __bf16 v16bf;
typedef __attribute__((ext_vector_type(8)))  float  v8f;

__device__ __forceinline__ float gelu_exact(float v) {
    return 0.5f * v * (1.0f + erff(v * 0.70710678118654752440f));
}

// ---------------------------------------------------------------- small utils
__global__ void moe_zero_f(float* p, int n) {
    int i = threadIdx.x;
    if (i < n) p[i] = 0.0f;
}

__global__ __launch_bounds__(256) void moe_zero16(uint4* p, long long n16) {
    long long i = (long long)blockIdx.x * 256 + threadIdx.x;
    if (i < n16) p[i] = make_uint4(0u, 0u, 0u, 0u);
}

// ---------------------------------------------------------------- router
// 8 waves per block, one token per wave.  Wg (8x1024) cached in LDS.
__global__ __launch_bounds__(256) void moe_router(
    const float* __restrict__ x, const float* __restrict__ Wg,
    int* __restrict__ topi, float* __restrict__ topv, float* __restrict__ imp)
{
    __shared__ float WgS[NE * DIM];
    __shared__ float impS[NE];
    int tid = threadIdx.x;
    if (tid < NE) impS[tid] = 0.0f;
    for (int i = tid; i < NE * DIM; i += 256) WgS[i] = Wg[i];
    __syncthreads();

    int lane = tid & 31, wave = tid >> 5;
    int t = blockIdx.x * 8 + wave;

    float acc[NE];
#pragma unroll
    for (int e = 0; e < NE; ++e) acc[e] = 0.0f;

    const float* xr = x + (size_t)t * DIM;
    for (int d = lane; d < DIM; d += 32) {
        float xv = xr[d];
#pragma unroll
        for (int e = 0; e < NE; ++e) acc[e] += xv * WgS[e * DIM + d];
    }
#pragma unroll
    for (int off = 16; off > 0; off >>= 1) {
#pragma unroll
        for (int e = 0; e < NE; ++e) acc[e] += __shfl_xor(acc[e], off, 32);
    }
    // softmax over E=8 (replicated on all lanes)
    float m = acc[0];
#pragma unroll
    for (int e = 1; e < NE; ++e) m = fmaxf(m, acc[e]);
    float g[NE], s = 0.0f;
#pragma unroll
    for (int e = 0; e < NE; ++e) { g[e] = expf(acc[e] - m); s += g[e]; }
#pragma unroll
    for (int e = 0; e < NE; ++e) g[e] /= s;
    // top-2, ties keep lowest index (matches lax.top_k)
    float v0 = -1.0f, v1 = -1.0f; int i0 = 0, i1 = 0;
#pragma unroll
    for (int e = 0; e < NE; ++e) {
        if (g[e] > v0)      { v1 = v0; i1 = i0; v0 = g[e]; i0 = e; }
        else if (g[e] > v1) { v1 = g[e]; i1 = e; }
    }
    if (lane == 0) {
        topi[t * 2] = i0; topi[t * 2 + 1] = i1;
        topv[t * 2] = v0; topv[t * 2 + 1] = v1;
    }
    if (lane < NE) atomicAdd(&impS[lane], g[lane]);
    __syncthreads();
    if (tid < NE) atomicAdd(&imp[tid], impS[tid]);
}

// ---------------------------------------------------------------- dispatch
// Single-wave deterministic FCFS scan in slot-major (k outer, t inner) order.
__global__ void moe_dispatch(const int* __restrict__ topi,
                             int* __restrict__ pos, float* __restrict__ tpe)
{
    int lane = threadIdx.x; // 32 threads
    __shared__ int base[NE];
    if (lane < NE) base[lane] = 0;
    __syncthreads();
    unsigned lt = (1u << lane) - 1u;
    for (int k = 0; k < TOPK; ++k) {
        for (int t0 = 0; t0 < TOK; t0 += 32) {
            int t = t0 + lane;
            int e = topi[t * TOPK + k];
            unsigned bal[NE];
#pragma unroll
            for (int ee = 0; ee < NE; ++ee)
                bal[ee] = (unsigned)__ballot(e == ee);
            int rank = base[e] + __popc(bal[e] & lt);
            pos[t * TOPK + k] = (rank < CAP) ? rank : CAP;
            __syncthreads();
            if (lane < NE) base[lane] += __popc(bal[lane]);
            __syncthreads();
        }
    }
    if (lane < NE) tpe[lane] = (float)((base[lane] < CAP) ? base[lane] : CAP);
}

// ---------------------------------------------------------------- combine weights
__global__ __launch_bounds__(256) void moe_weights(
    const float* __restrict__ topv, const int* __restrict__ pos,
    float* __restrict__ w)
{
    int t = blockIdx.x * 256 + threadIdx.x;
    float w0 = (pos[t * 2]     < CAP) ? topv[t * 2]     : 0.0f;
    float w1 = (pos[t * 2 + 1] < CAP) ? topv[t * 2 + 1] : 0.0f;
    float s = w0 + w1;
    if (s > 0.0f) { w0 /= s; w1 /= s; }
    w[t * 2] = w0; w[t * 2 + 1] = w1;
}

// ---------------------------------------------------------------- scatter to bf16 expert buffers
__global__ __launch_bounds__(128) void moe_scatter(
    const float* __restrict__ x, const int* __restrict__ topi,
    const int* __restrict__ pos, unsigned short* __restrict__ Xe_)
{
    int a = blockIdx.x;       // assignment id in [0, TOK*TOPK)
    int t = a >> 1, k = a & 1;
    int p = pos[t * 2 + k];
    if (p >= CAP) return;
    int e = topi[t * 2 + k];
    __bf16* Xe = (__bf16*)Xe_;
    const float* src = x + (size_t)t * DIM;
    __bf16* dst = Xe + ((size_t)e * CPAD + p) * DIM;
    int d0 = threadIdx.x * 8;
    float4 f0 = *(const float4*)(src + d0);
    float4 f1 = *(const float4*)(src + d0 + 4);
    __align__(16) __bf16 o[8];
    o[0] = (__bf16)f0.x; o[1] = (__bf16)f0.y; o[2] = (__bf16)f0.z; o[3] = (__bf16)f0.w;
    o[4] = (__bf16)f1.x; o[5] = (__bf16)f1.y; o[6] = (__bf16)f1.z; o[7] = (__bf16)f1.w;
    *(uint4*)(dst + d0) = *(const uint4*)o;
}

// ---------------------------------------------------------------- WMMA GEMM
// C[M,N] = A[M,K] (bf16, row-major) x B[K,N] (f32->bf16, row-major) + bias
// Block tile 128x128, K-step 32, 8 waves (2 in M x 4 in N), each wave 64x32
// (= 4x2 fragments of v_wmma_f32_16x16x32_bf16).
// LDS tiles are stored in the exact per-lane fragment layout of the ISA
// (16-bit A 16x32 / B 32x16 tables), so frag loads are 2x b128 per lane.
__global__ __launch_bounds__(256) void moe_gemm(
    const unsigned short* __restrict__ A_, const float* __restrict__ Bw,
    const float* __restrict__ bias, void* __restrict__ Out,
    int Kd, int Nd, int fuse /*1: +bias,gelu,bf16 out; 0: +bias,f32 out*/)
{
    __shared__ __align__(32) __bf16 As[8][32][16];
    __shared__ __align__(32) __bf16 Bs[8][32][16];

    const __bf16* A = (const __bf16*)A_;
    const int tid  = threadIdx.x;
    const int lane = tid & 31;
    const int wave = tid >> 5;
    const int wm = wave & 1;   // 0..1 : M half
    const int wn = wave >> 1;  // 0..3 : N quarter
    const int mBase = blockIdx.y * 128;
    const int nBase = blockIdx.x * 128;

    v8f acc[4][2];
    const v8f vzero = {0.f, 0.f, 0.f, 0.f, 0.f, 0.f, 0.f, 0.f};
#pragma unroll
    for (int i = 0; i < 4; ++i)
#pragma unroll
        for (int j = 0; j < 2; ++j) acc[i][j] = vzero;

    // A-loader mapping (per thread): row r = tid>>1, K-half h = tid&1
    const int ar = tid >> 1, ah = tid & 1;
    const int amt = ar >> 4, al0 = ar & 15;
    // B-loader mapping: col = tid&127, K-group g = tid>>7
    const int bcol = tid & 127, bg = tid >> 7;
    const int bnt = bcol >> 4, bl = (bcol & 15) + bg * 16;

    for (int k0 = 0; k0 < Kd; k0 += 32) {
        // ---- stage A tile (128x32 bf16) into frag-ready LDS ----
        {
            const __bf16* src = A + (size_t)(mBase + ar) * Kd + k0 + ah * 16;
            uint4 d0 = *(const uint4*)(src);      // K j..j+7 (rows l0)
            uint4 d1 = *(const uint4*)(src + 8);  // K j+8..j+15 (rows l0+16)
            *(uint4*)&As[amt][al0][ah * 8]      = d0;
            *(uint4*)&As[amt][al0 + 16][ah * 8] = d1;
        }
        // ---- stage B tile (32x128 f32 -> bf16) into frag-ready LDS ----
        {
            const float* src = Bw + (size_t)(k0 + bg * 8) * Nd + nBase + bcol;
            __align__(16) __bf16 tmp[16];
#pragma unroll
            for (int j = 0; j < 16; ++j) tmp[j] = (__bf16)src[(size_t)(j + (j & 8)) * Nd];
            *(uint4*)&Bs[bnt][bl][0] = *(const uint4*)&tmp[0];
            *(uint4*)&Bs[bnt][bl][8] = *(const uint4*)&tmp[8];
        }
        __syncthreads();

        v16bf af[4], bf[2];
#pragma unroll
        for (int i = 0; i < 4; ++i)
            af[i] = *(const v16bf*)&As[wm * 4 + i][lane][0];
#pragma unroll
        for (int j = 0; j < 2; ++j)
            bf[j] = *(const v16bf*)&Bs[wn * 2 + j][lane][0];

#pragma unroll
        for (int i = 0; i < 4; ++i)
#pragma unroll
            for (int j = 0; j < 2; ++j)
                acc[i][j] = __builtin_amdgcn_wmma_f32_16x16x32_bf16(
                    false, af[i], false, bf[j], (short)0, acc[i][j],
                    false, false);
        __syncthreads();
    }

    // ---- epilogue ----
    const int lmod = lane & 15, lhi = lane >> 4;
#pragma unroll
    for (int j = 0; j < 2; ++j) {
        int n = nBase + (wn * 2 + j) * 16 + lmod;
        float bv = bias[n];
#pragma unroll
        for (int i = 0; i < 4; ++i) {
            int mtop = mBase + (wm * 4 + i) * 16 + lhi * 8;
#pragma unroll
            for (int r = 0; r < 8; ++r) {
                float v = acc[i][j][r] + bv;
                size_t idx = (size_t)(mtop + r) * Nd + n;
                if (fuse) ((__bf16*)Out)[idx] = (__bf16)gelu_exact(v);
                else      ((float*)Out)[idx]  = v;
            }
        }
    }
}

// ---------------------------------------------------------------- combine
__global__ __launch_bounds__(256) void moe_combine(
    const float* __restrict__ Ye, const int* __restrict__ topi,
    const int* __restrict__ pos, const float* __restrict__ w,
    float* __restrict__ y)
{
    int t = blockIdx.x;
    int e0 = topi[t * 2], e1 = topi[t * 2 + 1];
    int p0 = pos[t * 2],  p1 = pos[t * 2 + 1];
    float w0 = w[t * 2],  w1 = w[t * 2 + 1];
    bool v0 = p0 < CAP, v1 = p1 < CAP;
    const float* r0 = Ye + ((size_t)e0 * CPAD + (v0 ? p0 : 0)) * DIM;
    const float* r1 = Ye + ((size_t)e1 * CPAD + (v1 ? p1 : 0)) * DIM;
    for (int d = threadIdx.x; d < DIM; d += 256) {
        float a = 0.0f;
        if (v0) a += w0 * r0[d];
        if (v1) a += w1 * r1[d];
        y[(size_t)t * DIM + d] = a;
    }
}

// ---------------------------------------------------------------- aux losses
__global__ void moe_losses(const float* __restrict__ imp,
                           const float* __restrict__ tpe,
                           float* __restrict__ out)
{
    if (threadIdx.x != 0) return;
    float m1 = 0.f, m2 = 0.f;
    for (int e = 0; e < NE; ++e) { m1 += imp[e]; m2 += tpe[e]; }
    m1 /= NE; m2 /= NE;
    float va = 0.f, vb = 0.f;
    for (int e = 0; e < NE; ++e) {
        float d1 = imp[e] - m1; va += d1 * d1;
        float d2 = tpe[e] - m2; vb += d2 * d2;
    }
    va /= NE; vb /= NE;
    float li = va / ((m1 + 1e-6f) * (m1 + 1e-6f));
    float ll = vb / ((m2 + 1e-6f) * (m2 + 1e-6f));
    out[0] = 0.5f * (li + ll);
    out[1] = ll;
}

// ================================================================ launch
extern "C" void kernel_launch(void* const* d_in, const int* in_sizes, int n_in,
                              void* d_out, int out_size, void* d_ws, size_t ws_size,
                              hipStream_t stream)
{
    const float* x  = (const float*)d_in[0];
    const float* Wg = (const float*)d_in[1];
    const float* W1 = (const float*)d_in[2];
    const float* b1 = (const float*)d_in[3];
    const float* W2 = (const float*)d_in[4];
    const float* b2 = (const float*)d_in[5];
    float* y = (float*)d_out;

    char* ws = (char*)d_ws;
    size_t off = 0;
    auto take = [&](size_t bytes) -> char* {
        char* p = ws + off;
        off = (off + bytes + 255) & ~(size_t)255;
        return p;
    };
    float*          imp  = (float*)take(NE * 4);
    float*          tpe  = (float*)take(NE * 4);
    int*            topi = (int*)take((size_t)TOK * TOPK * 4);
    float*          topv = (float*)take((size_t)TOK * TOPK * 4);
    int*            pos  = (int*)take((size_t)TOK * TOPK * 4);
    float*          wcb  = (float*)take((size_t)TOK * TOPK * 4);
    unsigned short* Xe   = (unsigned short*)take((size_t)NE * CPAD * DIM * 2);
    unsigned short* He   = (unsigned short*)take((size_t)CPAD * HID * 2);
    float*          Ye   = (float*)take((size_t)NE * CPAD * DIM * 4);

    moe_zero_f<<<1, 32, 0, stream>>>(imp, NE);
    long long n16 = (long long)NE * CPAD * DIM / 8; // uint4 count of Xe
    moe_zero16<<<(unsigned)((n16 + 255) / 256), 256, 0, stream>>>((uint4*)Xe, n16);

    moe_router<<<TOK / 8, 256, 0, stream>>>(x, Wg, topi, topv, imp);
    moe_dispatch<<<1, 32, 0, stream>>>(topi, pos, tpe);
    moe_weights<<<TOK / 256, 256, 0, stream>>>(topv, pos, wcb);
    moe_scatter<<<TOK * TOPK, 128, 0, stream>>>(x, topi, pos, Xe);

    for (int e = 0; e < NE; ++e) {
        moe_gemm<<<dim3(HID / 128, CPAD / 128), 256, 0, stream>>>(
            Xe + (size_t)e * CPAD * DIM, W1 + (size_t)e * DIM * HID,
            b1 + (size_t)e * HID, He, DIM, HID, 1);
        moe_gemm<<<dim3(DIM / 128, CPAD / 128), 256, 0, stream>>>(
            He, W2 + (size_t)e * HID * DIM,
            b2 + (size_t)e * DIM, Ye + (size_t)e * CPAD * DIM, HID, DIM, 0);
    }

    moe_combine<<<TOK, 256, 0, stream>>>(Ye, topi, pos, wcb, y);
    moe_losses<<<1, 32, 0, stream>>>(imp, tpe, y + (size_t)TOK * DIM);
}
